// FusedDecoder_25890062860905
// MI455X (gfx1250) — compile-verified, matrix operand order checked by reading, not run on hardware
//
#include <hip/hip_runtime.h>
#include <hip/hip_bf16.h>
#include <cstdint>

typedef __attribute__((ext_vector_type(16))) __bf16 v16bf;
typedef __attribute__((ext_vector_type(8)))  float  v8f;

#define BN_EPS 1e-5f

// ---------------------------------------------------------------------------
// CDNA5 async-load-to-LDS support (GLOBAL_LOAD_ASYNC_TO_LDS_B128, ASYNCcnt)
// ---------------------------------------------------------------------------
#if defined(__has_builtin)
#if __has_builtin(__builtin_amdgcn_global_load_async_to_lds_b128) && \
    __has_builtin(__builtin_amdgcn_s_wait_asynccnt)
#define HAS_ASYNC 1
#endif
#endif
#ifndef HAS_ASYNC
#define HAS_ASYNC 0
#endif

#if HAS_ASYNC
// Builtin signature (from clang diagnostic): params are pointers to a 16-byte
// int vector; src in AS1 (global, printed as __device__), dst in AS3 (LDS).
typedef int v4i_vs __attribute__((vector_size(16)));
// One instruction moves 16B per lane (512B/wave) from global to LDS.
// AS3 offset is the low-32-bits of the generic LDS pointer (ISA: LDS aperture
// addresses truncate to addr[31:0]).
__device__ __forceinline__ void async_copy16(const void* g, void* l) {
  __builtin_amdgcn_global_load_async_to_lds_b128(
      (__attribute__((address_space(1))) v4i_vs*)(uintptr_t)g,
      (__attribute__((address_space(3))) v4i_vs*)(uint32_t)(uintptr_t)l,
      0, 0);
}
#endif

// ---------------------------------------------------------------------------
// WMMA helpers (CDNA5 wave32, V_WMMA_F32_16X16X32_BF16)
// ---------------------------------------------------------------------------
__device__ __forceinline__ v8f wmma_bf16(v16bf a, v16bf b, v8f c) {
  // (neg_a, A, neg_b, B, c_mod, C, reuse_a, reuse_b)
  return __builtin_amdgcn_wmma_f32_16x16x32_bf16(false, a, false, b, (short)0, c,
                                                 false, false);
}

// K index within a 32-wide K chunk for element i of a 16-bit A/B fragment.
// Per ISA 16-bit 16x32 layout: lanes 0-15 hold K pairs {0..7,16..23},
// lanes 16-31 hold {8..15,24..31} (half = lane>>4).
__device__ __forceinline__ int kmap(int i, int half) {
  int j = i >> 1, p = i & 1;
  int base = (j < 4) ? (j * 2) : (16 + (j - 4) * 2);
  return base + p + half * 8;
}

// Load one 16x32 bf16 A fragment from a row-major [M][rowlen] weight matrix.
// K pairs are contiguous in memory -> compiler merges into wide loads.
__device__ __forceinline__ v16bf load_afrag(const __bf16* __restrict__ w,
                                            int rowlen, int m, int kbase, int half) {
  v16bf a;
  const __bf16* p = w + (size_t)m * rowlen + kbase + half * 8;
#pragma unroll
  for (int j = 0; j < 8; ++j) {
    int base = (j < 4) ? (j * 2) : (16 + (j - 4) * 2);
    a[2 * j]     = p[base];
    a[2 * j + 1] = p[base + 1];
  }
  return a;
}

// ---------------------------------------------------------------------------
// Prep: fold BN into bf16 weights / fp32 biases
// ---------------------------------------------------------------------------
__global__ void fold_w_kernel(const float* __restrict__ w, const float* __restrict__ g,
                              const float* __restrict__ b, const float* __restrict__ m,
                              const float* __restrict__ v, __bf16* __restrict__ wout,
                              float* __restrict__ bias, int Cin) {
  int i = blockIdx.x * blockDim.x + threadIdx.x;
  int n = 64 * Cin;
  if (i < n) {
    int o = i / Cin;
    float s = g[o] * rsqrtf(v[o] + BN_EPS);
    wout[i] = (__bf16)(w[i] * s);
  }
  if (i < 64) {
    float s = g[i] * rsqrtf(v[i] + BN_EPS);
    bias[i] = b[i] - m[i] * s;
  }
}

// Reorder dcn / offset weights into [M][K = tap*64 + c] bf16, fold final BN.
__global__ void prep_dcn_kernel(const float* __restrict__ dcn_w,   // [64][64][3][3]
                                const float* __restrict__ off_w,   // [27][64][3][3]
                                const float* __restrict__ gf, const float* __restrict__ bnb,
                                const float* __restrict__ mf, const float* __restrict__ vf,
                                __bf16* __restrict__ wd,   // [64][576]
                                __bf16* __restrict__ wo,   // [32][576] (rows 27..31 zero)
                                float* __restrict__ fscale, float* __restrict__ fbias) {
  int i = blockIdx.x * blockDim.x + threadIdx.x;
  if (i < 64 * 576) {
    int m = i / 576, r = i % 576, k = r >> 6, c = r & 63;
    wd[i] = (__bf16)dcn_w[(size_t)(m * 64 + c) * 9 + k];
  }
  if (i < 32 * 576) {
    int m = i / 576, r = i % 576, k = r >> 6, c = r & 63;
    wo[i] = (m < 27) ? (__bf16)off_w[(size_t)(m * 64 + c) * 9 + k] : (__bf16)0.0f;
  }
  if (i < 64) {
    float s = gf[i] * rsqrtf(vf[i] + BN_EPS);
    fscale[i] = s;
    fbias[i]  = bnb[i] - mf[i] * s;
  }
}

// ---------------------------------------------------------------------------
// 1x1 conv + folded BN as a WMMA GEMM: out[64, B*HW] = W[64,Cin] x in[Cin, B*HW]
// wave-per-tile: 16 pixels x 64 out channels (4 M-tiles).
// Async path: wave-private double-buffered LDS slabs filled by
// GLOBAL_LOAD_ASYNC_TO_LDS_B128, synchronized only with s_wait_asynccnt.
// ---------------------------------------------------------------------------
__global__ __launch_bounds__(256)
void conv1x1_wmma_kernel(const float* __restrict__ in, const __bf16* __restrict__ wf,
                         const float* __restrict__ bias, float* __restrict__ out,
                         int Cin, int HW, int ngroups_total) {
  int wave = threadIdx.x >> 5, lane = threadIdx.x & 31;
  int gid = blockIdx.x * 8 + wave;
  if (gid >= ngroups_total) gid = ngroups_total - 1;   // duplicate tail work (idempotent)
  int ngroups = HW / 16;
  int b   = gid / ngroups;
  int hw0 = (gid % ngroups) * 16;
  int half = lane >> 4, lq = lane & 15;
  v8f acc[4] = {};
  const float* inb = in + (size_t)b * Cin * HW + hw0;
  int nchunks = Cin >> 5;

#if HAS_ASYNC
  __shared__ float tile[8][2][32][16];   // per-wave double-buffered [K=32][N=16]
  // Prologue: stage chunk 0 (4 async b128 per lane covers the 2KB slab).
#pragma unroll
  for (int q = 0; q < 4; ++q) {
    int s = lane + 32 * q;               // 16B segment id, 0..127
    int row = s >> 2, cs = (s & 3) * 4;
    async_copy16(inb + (size_t)row * HW + cs, &tile[wave][0][row][cs]);
  }
  for (int ci = 0; ci < nchunks; ++ci) {
    int cur = ci & 1;
    if (ci + 1 < nchunks) {
      int c0n = (ci + 1) << 5;
#pragma unroll
      for (int q = 0; q < 4; ++q) {
        int s = lane + 32 * q;
        int row = s >> 2, cs = (s & 3) * 4;
        async_copy16(inb + (size_t)(c0n + row) * HW + cs, &tile[wave][cur ^ 1][row][cs]);
      }
      __builtin_amdgcn_s_wait_asynccnt(4);   // chunk ci landed, ci+1 in flight
    } else {
      __builtin_amdgcn_s_wait_asynccnt(0);
    }
    asm volatile("" ::: "memory");
    v16bf bfrag;
#pragma unroll
    for (int i = 0; i < 16; ++i)
      bfrag[i] = (__bf16)tile[wave][cur][kmap(i, half)][lq];
    int c0 = ci << 5;
#pragma unroll
    for (int mt = 0; mt < 4; ++mt) {
      v16bf a = load_afrag(wf, Cin, mt * 16 + lq, c0, half);
      acc[mt] = wmma_bf16(a, bfrag, acc[mt]);
    }
    asm volatile("" ::: "memory");
  }
#else
  __shared__ float tile1[8][32][16];
  for (int ci = 0; ci < nchunks; ++ci) {
    int c0 = ci << 5;
    __syncthreads();
    if (c0 + 32 < Cin)
      __builtin_prefetch(inb + (size_t)(c0 + 32) * HW, 0, 1);
#pragma unroll
    for (int j = 0; j < 16; ++j) {
      int idx = lane + 32 * j;
      int row = idx >> 4, col = idx & 15;
      tile1[wave][row][col] = inb[(size_t)(c0 + row) * HW + col];
    }
    __syncthreads();
    v16bf bfrag;
#pragma unroll
    for (int i = 0; i < 16; ++i)
      bfrag[i] = (__bf16)tile1[wave][kmap(i, half)][lq];
#pragma unroll
    for (int mt = 0; mt < 4; ++mt) {
      v16bf a = load_afrag(wf, Cin, mt * 16 + lq, c0, half);
      acc[mt] = wmma_bf16(a, bfrag, acc[mt]);
    }
  }
#endif

  float* outb = out + (size_t)b * 64 * HW + hw0 + lq;
#pragma unroll
  for (int mt = 0; mt < 4; ++mt)
#pragma unroll
    for (int r = 0; r < 8; ++r) {
      int o = mt * 16 + r + half * 8;
      outb[(size_t)o * HW] = acc[mt][r] + bias[o];
    }
}

// ---------------------------------------------------------------------------
// Bilinear (align_corners=True) upsample of 3 coarse scales + sum + ReLU.
// ---------------------------------------------------------------------------
__device__ __forceinline__ float bilin_ac(const float* __restrict__ p, int Hin, int Win,
                                          int y, int x) {
  float fy = (float)y * (float)(Hin - 1) / 159.0f;
  float fx = (float)x * (float)(Win - 1) / 159.0f;
  int y0 = (int)floorf(fy); y0 = y0 > Hin - 2 ? Hin - 2 : (y0 < 0 ? 0 : y0);
  int x0 = (int)floorf(fx); x0 = x0 > Win - 2 ? Win - 2 : (x0 < 0 ? 0 : x0);
  float ty = fy - (float)y0, tx = fx - (float)x0;
  const float* r0 = p + (size_t)y0 * Win + x0;
  float a = r0[0], bq = r0[1], c = r0[Win], d = r0[Win + 1];
  float top = a + (bq - a) * tx;
  float bot = c + (d - c) * tx;
  return top + (bot - top) * ty;
}

__global__ void combine_kernel(float* __restrict__ x, const float* __restrict__ s1,
                               const float* __restrict__ s2, const float* __restrict__ s3) {
  const int HW = 160 * 160;
  int idx = blockIdx.x * blockDim.x + threadIdx.x;
  if (idx >= 4 * 64 * HW) return;
  int xw = idx % 160;
  int yh = (idx / 160) % 160;
  int bc = idx / HW;
  float v = x[idx];
  v += bilin_ac(s1 + (size_t)bc * 80 * 80, 80, 80, yh, xw);
  v += bilin_ac(s2 + (size_t)bc * 40 * 40, 40, 40, yh, xw);
  v += bilin_ac(s3 + (size_t)bc * 20 * 20, 20, 20, yh, xw);
  x[idx] = fmaxf(v, 0.0f);
}

// ---------------------------------------------------------------------------
// Offset 3x3 conv (64 -> 27, padded to 32) as implicit-im2col WMMA GEMM.
// ---------------------------------------------------------------------------
__global__ __launch_bounds__(256)
void offconv_wmma_kernel(const float* __restrict__ x, const __bf16* __restrict__ wo,
                         const float* __restrict__ off_b, float* __restrict__ om) {
  const int H = 160, W = 160, HW = H * W, C = 64;
  int wave = threadIdx.x >> 5, lane = threadIdx.x & 31;
  int gid = blockIdx.x * 8 + wave;                 // 4 * 160 * 10 = 6400 groups
  int b   = gid / (H * (W / 16));
  int rem = gid % (H * (W / 16));
  int y   = rem / (W / 16);
  int x0  = (rem % (W / 16)) * 16;
  int half = lane >> 4, lq = lane & 15;
  v8f acc[2] = {};
  const float* xb = x + (size_t)b * C * HW;

  for (int kk = 0; kk < 9; ++kk) {
    int dy = kk / 3 - 1, dx = kk % 3 - 1;
    int yy = y + dy;
    bool yok = (yy >= 0) && (yy < H);
    int xx = x0 + lq + dx;
    bool xok = (xx >= 0) && (xx < W);
#pragma unroll
    for (int h = 0; h < 2; ++h) {
      int c0 = h * 32;
      v16bf bfrag;
#pragma unroll
      for (int i = 0; i < 16; ++i) {
        int K = kmap(i, half);
        float val = 0.0f;
        if (yok && xok) val = xb[(size_t)(c0 + K) * HW + yy * W + xx];
        bfrag[i] = (__bf16)val;
      }
      int Kbase = kk * 64 + c0;
      v16bf a0 = load_afrag(wo, 576, 0 + lq, Kbase, half);
      v16bf a1 = load_afrag(wo, 576, 16 + lq, Kbase, half);
      acc[0] = wmma_bf16(a0, bfrag, acc[0]);
      acc[1] = wmma_bf16(a1, bfrag, acc[1]);
    }
  }

  float* omb = om + (size_t)b * 27 * HW + (size_t)y * W + x0 + lq;
#pragma unroll
  for (int r = 0; r < 8; ++r) {
    int o = r + half * 8;
    omb[(size_t)o * HW] = acc[0][r] + off_b[o];
    int o2 = 16 + o;
    if (o2 < 27) omb[(size_t)o2 * HW] = acc[1][r] + off_b[o2];
  }
}

// ---------------------------------------------------------------------------
// DCNv2 main conv: D[64, pix] = Wd[64,576] x Val[576, pix], where
// Val[k*64+c, n] = mask[k,n] * bilinear(x[c], sample(k,n)); folded BN + ReLU.
// ---------------------------------------------------------------------------
__global__ __launch_bounds__(256)
void dcn_wmma_kernel(const float* __restrict__ x, const float* __restrict__ om,
                     const __bf16* __restrict__ wd, const float* __restrict__ fscale,
                     const float* __restrict__ fbias, float* __restrict__ out) {
  const int H = 160, W = 160, HW = H * W, C = 64;
  __shared__ float sm[8][9][16][5];   // [wave][tap][pixel][{y0f,x0f,ty,tx,mask}]
  int wave = threadIdx.x >> 5, lane = threadIdx.x & 31;
  int gid = blockIdx.x * 8 + wave;                 // 6400 groups
  int b   = gid / (H * (W / 16));
  int rem = gid % (H * (W / 16));
  int y   = rem / (W / 16);
  int x0  = (rem % (W / 16)) * 16;
  int half = lane >> 4, lq = lane & 15;

  const float* omb = om + (size_t)b * 27 * HW + (size_t)y * W + x0;
  for (int it = lane; it < 144; it += 32) {
    int k = it / 16, n = it % 16;
    float dy = omb[(size_t)(2 * k) * HW + n];
    float dx = omb[(size_t)(2 * k + 1) * HW + n];
    float mk = 1.0f / (1.0f + __expf(-omb[(size_t)(18 + k) * HW + n]));
    float sy = (float)(y + k / 3 - 1) + dy;
    float sx = (float)(x0 + n + k % 3 - 1) + dx;
    float y0f = floorf(sy), x0f = floorf(sx);
    sm[wave][k][n][0] = y0f;
    sm[wave][k][n][1] = x0f;
    sm[wave][k][n][2] = sy - y0f;
    sm[wave][k][n][3] = sx - x0f;
    sm[wave][k][n][4] = mk;
  }
  __syncthreads();

  v8f acc[4] = {};
  const float* xb = x + (size_t)b * C * HW;
  for (int kk = 0; kk < 9; ++kk) {
    float y0f = sm[wave][kk][lq][0];
    float x0f = sm[wave][kk][lq][1];
    float ty  = sm[wave][kk][lq][2];
    float tx  = sm[wave][kk][lq][3];
    float mk  = sm[wave][kk][lq][4];
    int iy = (int)y0f, ix = (int)x0f;
    bool y0ok = (iy >= 0) && (iy < H);
    bool y1ok = (iy + 1 >= 0) && (iy + 1 < H);
    bool x0ok = (ix >= 0) && (ix < W);
    bool x1ok = (ix + 1 >= 0) && (ix + 1 < W);
#pragma unroll
    for (int h = 0; h < 2; ++h) {
      int c0 = h * 32;
      v16bf bfrag;
#pragma unroll
      for (int i = 0; i < 16; ++i) {
        int c = c0 + kmap(i, half);
        const float* xc = xb + (size_t)c * HW;
        float v00 = 0.f, v01 = 0.f, v10 = 0.f, v11 = 0.f;
        if (y0ok && x0ok) v00 = xc[iy * W + ix];
        if (y0ok && x1ok) v01 = xc[iy * W + ix + 1];
        if (y1ok && x0ok) v10 = xc[(iy + 1) * W + ix];
        if (y1ok && x1ok) v11 = xc[(iy + 1) * W + ix + 1];
        float top = v00 + (v01 - v00) * tx;
        float bot = v10 + (v11 - v10) * tx;
        bfrag[i] = (__bf16)((top + (bot - top) * ty) * mk);
      }
      int Kbase = kk * 64 + c0;
#pragma unroll
      for (int mt = 0; mt < 4; ++mt) {
        v16bf a = load_afrag(wd, 576, mt * 16 + lq, Kbase, half);
        acc[mt] = wmma_bf16(a, bfrag, acc[mt]);
      }
    }
  }

  float* ob = out + (size_t)b * 64 * HW + (size_t)y * W + x0 + lq;
#pragma unroll
  for (int mt = 0; mt < 4; ++mt)
#pragma unroll
    for (int r = 0; r < 8; ++r) {
      int o = mt * 16 + r + half * 8;
      ob[(size_t)o * HW] = fmaxf(acc[mt][r] * fscale[o] + fbias[o], 0.0f);
    }
}

// ---------------------------------------------------------------------------
// Host launcher
// ---------------------------------------------------------------------------
extern "C" void kernel_launch(void* const* d_in, const int* in_sizes, int n_in,
                              void* d_out, int out_size, void* d_ws, size_t ws_size,
                              hipStream_t stream) {
  (void)in_sizes; (void)n_in; (void)out_size; (void)ws_size;

  const float* f0 = (const float*)d_in[0];
  const float* f1 = (const float*)d_in[1];
  const float* f2 = (const float*)d_in[2];
  const float* f3 = (const float*)d_in[3];
  const float *w[4], *g[4], *bb[4], *m[4], *v[4];
  for (int i = 0; i < 4; ++i) {
    w[i]  = (const float*)d_in[4 + i * 5 + 0];
    g[i]  = (const float*)d_in[4 + i * 5 + 1];
    bb[i] = (const float*)d_in[4 + i * 5 + 2];
    m[i]  = (const float*)d_in[4 + i * 5 + 3];
    v[i]  = (const float*)d_in[4 + i * 5 + 4];
  }
  const float* off_w = (const float*)d_in[24];
  const float* off_b = (const float*)d_in[25];
  const float* dcn_w = (const float*)d_in[26];
  const float* gf = (const float*)d_in[27];
  const float* bf = (const float*)d_in[28];
  const float* mf = (const float*)d_in[29];
  const float* vf = (const float*)d_in[30];

  char* ws = (char*)d_ws;
  size_t off = 0;
  auto alloc = [&](size_t bytes) -> void* {
    void* p = ws + off;
    off += (bytes + 255) & ~(size_t)255;
    return p;
  };

  __bf16* wf0 = (__bf16*)alloc(64 * 64 * sizeof(__bf16));
  __bf16* wf1 = (__bf16*)alloc(64 * 128 * sizeof(__bf16));
  __bf16* wf2 = (__bf16*)alloc(64 * 256 * sizeof(__bf16));
  __bf16* wf3 = (__bf16*)alloc(64 * 512 * sizeof(__bf16));
  float*  bias = (float*)alloc(4 * 64 * sizeof(float));
  __bf16* wd  = (__bf16*)alloc(64 * 576 * sizeof(__bf16));
  __bf16* wo  = (__bf16*)alloc(32 * 576 * sizeof(__bf16));
  float*  fsc = (float*)alloc(64 * sizeof(float));
  float*  fbi = (float*)alloc(64 * sizeof(float));
  float*  s1   = (float*)alloc((size_t)4 * 64 * 80 * 80 * sizeof(float));
  float*  s2   = (float*)alloc((size_t)4 * 64 * 40 * 40 * sizeof(float));
  float*  s3   = (float*)alloc((size_t)4 * 64 * 20 * 20 * sizeof(float));
  float*  xbuf = (float*)alloc((size_t)4 * 64 * 160 * 160 * sizeof(float));
  float*  om   = (float*)alloc((size_t)4 * 27 * 160 * 160 * sizeof(float));

  __bf16* wfs[4] = {wf0, wf1, wf2, wf3};
  const int cins[4] = {64, 128, 256, 512};
  for (int i = 0; i < 4; ++i) {
    int nthr = 64 * cins[i];
    fold_w_kernel<<<(nthr + 255) / 256, 256, 0, stream>>>(
        w[i], g[i], bb[i], m[i], v[i], wfs[i], bias + i * 64, cins[i]);
  }
  prep_dcn_kernel<<<144, 256, 0, stream>>>(dcn_w, off_w, gf, bf, mf, vf, wd, wo, fsc, fbi);

  // 1x1 conv + BN per scale (WMMA GEMMs with async-to-LDS double buffering)
  conv1x1_wmma_kernel<<<800, 256, 0, stream>>>(f0, wf0, bias + 0,   xbuf, 64,  25600, 6400);
  conv1x1_wmma_kernel<<<200, 256, 0, stream>>>(f1, wf1, bias + 64,  s1,   128, 6400,  1600);
  conv1x1_wmma_kernel<<<50,  256, 0, stream>>>(f2, wf2, bias + 128, s2,   256, 1600,  400);
  conv1x1_wmma_kernel<<<13,  256, 0, stream>>>(f3, wf3, bias + 192, s3,   512, 400,   100);

  // upsample + sum + relu (in place on xbuf)
  combine_kernel<<<25600, 256, 0, stream>>>(xbuf, s1, s2, s3);

  // offset/mask conv (WMMA), then deformable conv + final BN + relu (WMMA)
  offconv_wmma_kernel<<<800, 256, 0, stream>>>(xbuf, wo, off_b, om);
  dcn_wmma_kernel<<<800, 256, 0, stream>>>(xbuf, om, wd, fsc, fbi, (float*)d_out);
}
